// GainAKT4_38087769980990
// MI455X (gfx1250) — compile-verified, matrix operand order checked by reading, not run on hardware
//
#include <hip/hip_runtime.h>
#include <hip/hip_bf16.h>
#include <stdint.h>

#define B_    32
#define SEQ_  512
#define D_    256
#define H_    4
#define DK_   64
#define NUMC_ 2000
#define DFF_  512
#define M_    (B_*SEQ_)   /* 16384 rows */

typedef __attribute__((ext_vector_type(16))) __bf16 v16bf;
typedef __attribute__((ext_vector_type(8)))  __bf16 v8bf;
typedef __attribute__((ext_vector_type(8)))  float  v8f;

__device__ __forceinline__ __bf16 f2bf(float f) {
  union { float f; uint32_t u; } v; v.f = f;
  uint32_t r = v.u + 0x7fffu + ((v.u >> 16) & 1u);
  uint16_t h = (uint16_t)(r >> 16);
  __bf16 o; __builtin_memcpy(&o, &h, 2); return o;
}
__device__ __forceinline__ float bf2f(__bf16 b) {
  uint16_t h; __builtin_memcpy(&h, &b, 2);
  union { uint32_t u; float f; } v; v.u = ((uint32_t)h) << 16; return v.f;
}
// Build a 16x32 bf16 A/B fragment for v_wmma_f32_16x16x32_bf16.
// Per ISA layout: lanes 0-15 hold K=[0..7] and [16..23], lanes 16-31 hold
// K=[8..15] and [24..31] -> two contiguous 8-element (16B) chunks per lane.
__device__ __forceinline__ v16bf make_frag(const __bf16* p, int lh) {
  v8bf lo = *(const v8bf*)(p + lh * 8);
  v8bf hi = *(const v8bf*)(p + lh * 8 + 16);
  v16bf f;
#pragma unroll
  for (int i = 0; i < 8; i++) { f[i] = lo[i]; f[i + 8] = hi[i]; }
  return f;
}
#define WMMA_BF16(a, b, c) \
  __builtin_amdgcn_wmma_f32_16x16x32_bf16(false, (a), false, (b), (short)0, (c), false, false)

// ---------------------------------------------------------------- embeddings
__global__ void embed_kernel(const int* __restrict__ q, const int* __restrict__ r,
                             const float* __restrict__ ce, const float* __restrict__ ve,
                             const float* __restrict__ pe,
                             float* __restrict__ ctx, float* __restrict__ val) {
  size_t total = (size_t)M_ * D_;
  for (size_t i = (size_t)blockIdx.x * blockDim.x + threadIdx.x; i < total;
       i += (size_t)gridDim.x * blockDim.x) {
    int bl = (int)(i / D_); int d = (int)(i % D_);
    int l = bl % SEQ_;
    int tok = q[bl] + NUMC_ * r[bl];
    float p = pe[(size_t)l * D_ + d];
    ctx[i] = ce[(size_t)tok * D_ + d] + p;
    val[i] = ve[(size_t)tok * D_ + d] + p;
  }
}

// -------------------------------------------------------------- conversions
__global__ void conv_bf16_kernel(const float* __restrict__ src, __bf16* __restrict__ dst, size_t n) {
  for (size_t i = (size_t)blockIdx.x * blockDim.x + threadIdx.x; i < n;
       i += (size_t)gridDim.x * blockDim.x)
    dst[i] = f2bf(src[i]);
}
// weight f32[K,N] -> bf16[Kpad,Npad], zero-filled outside [K,N]
__global__ void convw_kernel(const float* __restrict__ W, __bf16* __restrict__ dst,
                             int K, int N, int Kpad, int Npad) {
  size_t n = (size_t)Kpad * Npad;
  for (size_t i = (size_t)blockIdx.x * blockDim.x + threadIdx.x; i < n;
       i += (size_t)gridDim.x * blockDim.x) {
    int k = (int)(i / Npad); int c = (int)(i % Npad);
    dst[i] = (k < K && c < N) ? f2bf(W[(size_t)k * N + c]) : f2bf(0.f);
  }
}

// ------------------------------------------------------------------- GEMM
// C[M,N] = A[M,K](bf16, stride lda) x B[Kpad,Npad](bf16, stride ldb) + bias.
// 256 threads = 8 waves; block tile 128x64; each wave: 16 rows x 64 cols.
// ldb is a multiple of 64 so all B loads are unguarded 16B vectors.
__global__ __launch_bounds__(256) void gemm_bf16_kernel(
    const __bf16* __restrict__ A, int lda, const __bf16* __restrict__ Bw, int ldb,
    const float* __restrict__ bias, float* __restrict__ Cf, __bf16* __restrict__ Cb,
    int N, int K, int relu) {
  __shared__ __bf16 As[128][40] __attribute__((aligned(16)));
  __shared__ __bf16 Bs[64][40]  __attribute__((aligned(16)));  // transposed [n][k]
  int tid = threadIdx.x;
  int wave = tid >> 5, lane = tid & 31;
  int lh = (lane >> 4) & 1, l16 = lane & 15;
  int m0 = blockIdx.x * 128;
  int n0 = blockIdx.y * 64;

  const v8f vzero = {};
  v8f acc0 = vzero, acc1 = vzero, acc2 = vzero, acc3 = vzero;

  // per-thread staging coordinates
  int arow = tid >> 1, akc = (tid & 1) * 16;          // A: 16 elems/thread
  int bkk = tid >> 3, bnn = (tid & 7) * 8;            // B: 8 elems/thread
  const __bf16* aptr = A + (size_t)(m0 + arow) * lda + akc;
  const __bf16* bptr = Bw + (size_t)bkk * ldb + n0 + bnn;

  for (int k0 = 0; k0 < K; k0 += 32) {
    {   // stage A tile 128x32 (two unguarded 16B loads)
      v8bf a0 = *(const v8bf*)(aptr + k0);
      v8bf a1 = *(const v8bf*)(aptr + k0 + 8);
      *(v8bf*)(&As[arow][akc])     = a0;
      *(v8bf*)(&As[arow][akc + 8]) = a1;
    }
    {   // stage B tile 32x64 transposed into Bs[n][k] (one unguarded 16B load)
      v8bf x = *(const v8bf*)(bptr + (size_t)k0 * ldb);
#pragma unroll
      for (int i = 0; i < 8; i++) Bs[bnn + i][bkk] = x[i];
    }
    if (k0 + 32 < K) {   // prefetch next K-step tiles
      __builtin_prefetch(aptr + k0 + 32, 0, 1);
      __builtin_prefetch(bptr + (size_t)(k0 + 32) * ldb, 0, 1);
    }
    __syncthreads();
    v16bf af = make_frag(&As[wave * 16 + l16][0], lh);
    acc0 = WMMA_BF16(af, make_frag(&Bs[ 0 + l16][0], lh), acc0);
    acc1 = WMMA_BF16(af, make_frag(&Bs[16 + l16][0], lh), acc1);
    acc2 = WMMA_BF16(af, make_frag(&Bs[32 + l16][0], lh), acc2);
    acc3 = WMMA_BF16(af, make_frag(&Bs[48 + l16][0], lh), acc3);
    __syncthreads();
  }
  // epilogue: C layout -> row = vgpr idx + 8*lanehalf, col = lane&15
#pragma unroll
  for (int ns = 0; ns < 4; ns++) {
    v8f a = (ns == 0) ? acc0 : (ns == 1) ? acc1 : (ns == 2) ? acc2 : acc3;
    int col = n0 + ns * 16 + l16;
    if (col < N) {
      float bv = bias ? bias[col] : 0.f;
#pragma unroll
      for (int rr = 0; rr < 8; rr++) {
        size_t row = (size_t)(m0 + wave * 16 + rr + 8 * lh);
        float v = a[rr] + bv;
        if (relu) v = fmaxf(v, 0.f);
        if (Cf) Cf[row * N + col] = v;
        if (Cb) Cb[row * N + col] = f2bf(v);
      }
    }
  }
}

// -------------------------------------------------- flash attention (causal)
// grid (L/64, H, B), block 128 = 4 waves; wave handles 16 query rows.
__global__ __launch_bounds__(128) void attn_kernel(
    const __bf16* __restrict__ Qb, const __bf16* __restrict__ Kb,
    const __bf16* __restrict__ Vb, float* __restrict__ Out) {
  __shared__ __bf16 Vs[64][40]     __attribute__((aligned(16)));  // [d][key]
  __shared__ __bf16 Ps[4][16][40]  __attribute__((aligned(16)));  // per-wave P
  int qblk = blockIdx.x, h = blockIdx.y, b = blockIdx.z;
  int tid = threadIdx.x, wave = tid >> 5, lane = tid & 31;
  int lh = (lane >> 4) & 1, l16 = lane & 15;
  int q0w = qblk * 64 + wave * 16;
  const __bf16* Qh = Qb + (size_t)b * SEQ_ * D_ + h * DK_;
  const __bf16* Kh = Kb + (size_t)b * SEQ_ * D_ + h * DK_;
  const __bf16* Vh = Vb + (size_t)b * SEQ_ * D_ + h * DK_;

  int qrow = q0w + l16;
  v16bf qa0 = make_frag(Qh + (size_t)qrow * D_, lh);
  v16bf qa1 = make_frag(Qh + (size_t)qrow * D_ + 32, lh);

  float mrow[8], lrow[8];
#pragma unroll
  for (int i = 0; i < 8; i++) { mrow[i] = -1e30f; lrow[i] = 0.f; }
  const v8f vzero = {};
  v8f o0 = vzero, o1 = vzero, o2 = vzero, o3 = vzero;

  int vkk = tid >> 2, vdd = (tid & 3) * 16;    // V staging coords
  int nsteps = 2 * qblk + 2;  // keys 0 .. qblk*64+63, step 32
  for (int s = 0; s < nsteps; s++) {
    int k0 = s * 32;
    {   // stage V tile [32 keys x 64 d] transposed into Vs[d][key]
      const __bf16* src = Vh + (size_t)(k0 + vkk) * D_ + vdd;
      v8bf x0 = *(const v8bf*)(src);
      v8bf x1 = *(const v8bf*)(src + 8);
#pragma unroll
      for (int i = 0; i < 8; i++) {
        Vs[vdd + i][vkk]     = x0[i];
        Vs[vdd + 8 + i][vkk] = x1[i];
      }
    }
    __syncthreads();

    v8f s0 = vzero, s1 = vzero;   // S tiles: keys [k0..k0+15], [k0+16..k0+31]
#pragma unroll
    for (int dc = 0; dc < 2; dc++) {
      v16bf qa = dc ? qa1 : qa0;
      v16bf kf0 = make_frag(Kh + (size_t)(k0 + l16) * D_ + dc * 32, lh);
      s0 = WMMA_BF16(qa, kf0, s0);
      v16bf kf1 = make_frag(Kh + (size_t)(k0 + 16 + l16) * D_ + dc * 32, lh);
      s1 = WMMA_BF16(qa, kf1, s1);
    }
    int key0 = k0 + l16, key1 = k0 + 16 + l16;
#pragma unroll
    for (int rr = 0; rr < 8; rr++) {
      int qq = q0w + rr + 8 * lh;
      float a0 = (key0 <= qq) ? s0[rr] * 0.125f : -1e30f;
      float a1 = (key1 <= qq) ? s1[rr] * 0.125f : -1e30f;
      float rm = fmaxf(a0, a1);
#pragma unroll
      for (int off = 8; off >= 1; off >>= 1) rm = fmaxf(rm, __shfl_xor(rm, off, 32));
      float mn = fmaxf(mrow[rr], rm);
      float corr = __expf(mrow[rr] - mn);
      float p0 = __expf(a0 - mn);
      float p1 = __expf(a1 - mn);
      float rs = p0 + p1;
#pragma unroll
      for (int off = 8; off >= 1; off >>= 1) rs += __shfl_xor(rs, off, 32);
      lrow[rr] = lrow[rr] * corr + rs;
      mrow[rr] = mn;
      o0[rr] *= corr; o1[rr] *= corr; o2[rr] *= corr; o3[rr] *= corr;
      Ps[wave][rr + 8 * lh][l16]      = f2bf(p0);
      Ps[wave][rr + 8 * lh][16 + l16] = f2bf(p1);
    }
    __syncthreads();
    v16bf pa = make_frag(&Ps[wave][l16][0], lh);   // P as A-fragment
    o0 = WMMA_BF16(pa, make_frag(&Vs[ 0 + l16][0], lh), o0);
    o1 = WMMA_BF16(pa, make_frag(&Vs[16 + l16][0], lh), o1);
    o2 = WMMA_BF16(pa, make_frag(&Vs[32 + l16][0], lh), o2);
    o3 = WMMA_BF16(pa, make_frag(&Vs[48 + l16][0], lh), o3);
    __syncthreads();
  }
  float* Oh = Out + (size_t)b * SEQ_ * D_ + h * DK_;
#pragma unroll
  for (int rr = 0; rr < 8; rr++) {
    float invl = 1.f / lrow[rr];
    size_t rbase = (size_t)(q0w + rr + 8 * lh) * D_;
    Oh[rbase +  0 + l16] = o0[rr] * invl;
    Oh[rbase + 16 + l16] = o1[rr] * invl;
    Oh[rbase + 32 + l16] = o2[rr] * invl;
    Oh[rbase + 48 + l16] = o3[rr] * invl;
  }
}

// ---------------------------------------------------- residual + layernorm
__global__ __launch_bounds__(256) void ln_kernel(const float* __restrict__ X,
                                                 const float* __restrict__ Aa,
                                                 const float* __restrict__ g,
                                                 const float* __restrict__ bb,
                                                 float* __restrict__ Y) {
  int wave = threadIdx.x >> 5, lane = threadIdx.x & 31;
  size_t row = (size_t)blockIdx.x * 8 + wave;
  const float* x = X + row * D_;
  const float* a = Aa + row * D_;
  float v[8]; float s = 0.f;
#pragma unroll
  for (int i = 0; i < 8; i++) { v[i] = x[lane + 32 * i] + a[lane + 32 * i]; s += v[i]; }
#pragma unroll
  for (int off = 16; off >= 1; off >>= 1) s += __shfl_xor(s, off, 32);
  float mean = s * (1.f / D_);
  float qv = 0.f;
#pragma unroll
  for (int i = 0; i < 8; i++) { float d = v[i] - mean; qv += d * d; }
#pragma unroll
  for (int off = 16; off >= 1; off >>= 1) qv += __shfl_xor(qv, off, 32);
  float inv = rsqrtf(qv * (1.f / D_) + 1e-5f);
  float* y = Y + row * D_;
#pragma unroll
  for (int i = 0; i < 8; i++) {
    int c = lane + 32 * i;
    y[c] = (v[i] - mean) * inv * g[c] + bb[c];
  }
}

// ---------------------------------------------------------- head cat builders
__global__ void cat_head_kernel(const float* __restrict__ h1, const float* __restrict__ v1,
                                const float* __restrict__ skill, const int* __restrict__ qry,
                                __bf16* __restrict__ outp) {
  size_t total = (size_t)M_ * 768;
  for (size_t i = (size_t)blockIdx.x * blockDim.x + threadIdx.x; i < total;
       i += (size_t)gridDim.x * blockDim.x) {
    int bl = (int)(i / 768); int c = (int)(i % 768);
    float v;
    if (c < 256)      v = h1[(size_t)bl * 256 + c];
    else if (c < 512) v = v1[(size_t)bl * 256 + (c - 256)];
    else              v = skill[(size_t)qry[bl] * 256 + (c - 512)];
    outp[i] = f2bf(v);
  }
}
__global__ void cat_m_kernel(const float* __restrict__ h1, const float* __restrict__ v1,
                             const float* __restrict__ kc, const int* __restrict__ qry,
                             __bf16* __restrict__ outp) {
  size_t total = (size_t)M_ * 544;   // 513 padded to 544
  for (size_t i = (size_t)blockIdx.x * blockDim.x + threadIdx.x; i < total;
       i += (size_t)gridDim.x * blockDim.x) {
    int bl = (int)(i / 544); int c = (int)(i % 544);
    float v;
    if (c < 256)       v = h1[(size_t)bl * 256 + c];
    else if (c < 512)  v = v1[(size_t)bl * 256 + (c - 256)];
    else if (c == 512) v = kc[(size_t)bl * NUMC_ + qry[bl]];
    else               v = 0.f;
    outp[i] = f2bf(v);
  }
}

// -------------------------------------- final 256->1 dot + sigmoid per row
__global__ __launch_bounds__(256) void dot_head_kernel(const __bf16* __restrict__ X,
                                                       const float* __restrict__ W,
                                                       const float* __restrict__ bsc,
                                                       float* __restrict__ sig_out,
                                                       float* __restrict__ logit_out) {
  int wave = threadIdx.x >> 5, lane = threadIdx.x & 31;
  size_t row = (size_t)blockIdx.x * 8 + wave;
  const __bf16* x = X + row * 256;
  float s = 0.f;
#pragma unroll
  for (int i = 0; i < 8; i++) { int c = lane + 32 * i; s += bf2f(x[c]) * W[c]; }
#pragma unroll
  for (int off = 16; off >= 1; off >>= 1) s += __shfl_xor(s, off, 32);
  if (lane == 0) {
    float lg = s + bsc[0];
    logit_out[row] = lg;
    sig_out[row] = 1.f / (1.f + __expf(-lg));
  }
}

// ---------------------------------- in-place exp+cumsum+clip over L per (b,c)
__global__ void cumsum_kernel(float* __restrict__ kc, const float* __restrict__ scale) {
  int idx = blockIdx.x * blockDim.x + threadIdx.x;
  if (idx >= B_ * NUMC_) return;
  int b = idx / NUMC_, c = idx % NUMC_;
  float sc = scale[0];
  float s = 0.f;
  float* p = kc + (size_t)b * SEQ_ * NUMC_ + c;
  for (int l = 0; l < SEQ_; l++) {
    s += __expf(p[0] + sc);
    p[0] = fminf(s, 1.0f);
    p += NUMC_;
  }
}

// ======================================================================
extern "C" void kernel_launch(void* const* d_in, const int* in_sizes, int n_in,
                              void* d_out, int out_size, void* d_ws, size_t ws_size,
                              hipStream_t stream) {
  (void)in_sizes; (void)n_in; (void)out_size; (void)ws_size;
  const int* q   = (const int*)d_in[0];
  const int* r   = (const int*)d_in[1];
  const int* qry = (const int*)d_in[2];
  const float* ctx_emb   = (const float*)d_in[3];
  const float* val_emb   = (const float*)d_in[4];
  const float* skill_emb = (const float*)d_in[5];
  const float* pos_emb   = (const float*)d_in[6];
  auto P = [&](int i) { return (const float*)d_in[i]; };

  char* ws = (char*)d_ws;
  float*  ctx  = (float*)(ws + 0);           // M*256 f32
  float*  val  = (float*)(ws + 16777216);    // M*256 f32
  float*  aout = (float*)(ws + 33554432);    // M*256 f32 (attn-pre / attn / ffn out)
  __bf16* xbf  = (__bf16*)(ws + 50331648);   // M*768 bf16 activation staging
  __bf16* ybf  = (__bf16*)(ws + 75497472);   // M*512 bf16
  __bf16* qbf  = (__bf16*)(ws + 92274688);   // M*256 bf16
  __bf16* kbf  = (__bf16*)(ws + 100663296);  // M*256 bf16
  __bf16* vbf  = (__bf16*)(ws + 109051904);  // M*256 bf16
  __bf16* wbf  = (__bf16*)(ws + 117440512);  // up to 512x2048 bf16 weights (2 MB)

  float* out      = (float*)d_out;
  float* o_bce    = out;
  float* o_mast   = out + 16384;
  float* o_kc     = out + 32768;                       // [B,L,2000]
  float* o_logit  = out + 32768 + (size_t)M_ * NUMC_;
  float* o_mlogit = o_logit + 16384;

  auto conv = [&](const float* src, __bf16* dst, size_t n) {
    conv_bf16_kernel<<<4096, 256, 0, stream>>>(src, dst, n);
  };
  auto convW = [&](const float* Wsrc, int K, int N, int Kpad) {
    int Npad = (N + 63) & ~63;
    size_t n = (size_t)Kpad * Npad;
    convw_kernel<<<(unsigned)((n + 255) / 256), 256, 0, stream>>>(Wsrc, wbf, K, N, Kpad, Npad);
  };
  auto gemm = [&](const __bf16* A, int lda, const float* bias,
                  float* Cf, __bf16* Cb, int N, int K, int relu) {
    int Npad = (N + 63) & ~63;
    dim3 g(M_ / 128, (N + 63) / 64);
    gemm_bf16_kernel<<<g, 256, 0, stream>>>(A, lda, wbf, Npad, bias, Cf, Cb, N, K, relu);
  };
  auto lnorm = [&](float* X, float* Aa, const float* g, const float* bb, float* Y) {
    ln_kernel<<<M_ / 8, 256, 0, stream>>>(X, Aa, g, bb, Y);
  };

  // embeddings
  embed_kernel<<<4096, 256, 0, stream>>>(q, r, ctx_emb, val_emb, pos_emb, ctx, val);

  // encoder blocks
  for (int blk = 0; blk < 4; blk++) {
    int wi = 7 + blk * 18;
    // Q = ctx @ Wq + bq ; K = ctx @ Wk + bk (bf16 outputs)
    conv(ctx, xbf, (size_t)M_ * 256);
    convW(P(wi + 0), 256, 256, 256);
    gemm(xbf, 256, P(wi + 1), nullptr, qbf, 256, 256, 0);
    convW(P(wi + 2), 256, 256, 256);
    gemm(xbf, 256, P(wi + 3), nullptr, kbf, 256, 256, 0);
    // V = val @ Wv + bv
    conv(val, xbf, (size_t)M_ * 256);
    convW(P(wi + 4), 256, 256, 256);
    gemm(xbf, 256, P(wi + 5), nullptr, vbf, 256, 256, 0);
    // causal flash attention -> aout (f32, pre-projection)
    attn_kernel<<<dim3(SEQ_ / 64, H_, B_), 128, 0, stream>>>(qbf, kbf, vbf, aout);
    // a = attn @ Wo + bo
    conv(aout, xbf, (size_t)M_ * 256);
    convW(P(wi + 6), 256, 256, 256);
    gemm(xbf, 256, P(wi + 7), aout, nullptr, 256, 256, 0);
    // residual layernorms
    lnorm(ctx, aout, P(wi + 8),  P(wi + 9),  ctx);
    lnorm(val, aout, P(wi + 10), P(wi + 11), val);
    // FFN
    conv(ctx, xbf, (size_t)M_ * 256);
    convW(P(wi + 14), 256, 512, 256);
    gemm(xbf, 256, P(wi + 15), nullptr, ybf, 512, 256, 1);
    convW(P(wi + 16), 512, 256, 512);
    gemm(ybf, 512, P(wi + 17), aout, nullptr, 256, 512, 0);
    lnorm(ctx, aout, P(wi + 12), P(wi + 13), ctx);
  }

  // prediction head: cat[h1,v1,skill] -> 512 -> 256 -> 1
  cat_head_kernel<<<4096, 256, 0, stream>>>(ctx, val, skill_emb, qry, xbf);
  convW(P(79), 768, 512, 768);
  gemm(xbf, 768, P(80), nullptr, ybf, 512, 768, 1);
  convW(P(81), 512, 256, 512);
  gemm(ybf, 512, P(82), nullptr, qbf, 256, 512, 1);
  dot_head_kernel<<<M_ / 8, 256, 0, stream>>>(qbf, P(83), P(84), o_bce, o_logit);

  // mastery: li = relu(h1@W1+b1)@W2+b2 written straight into kc region
  conv(ctx, xbf, (size_t)M_ * 256);
  convW(P(85), 256, 512, 256);
  gemm(xbf, 256, P(86), nullptr, ybf, 512, 256, 1);
  convW(P(87), 512, NUMC_, 512);
  gemm(ybf, 512, P(88), o_kc, nullptr, NUMC_, 512, 0);
  // kc = clip(cumsum(exp(li + inc_scale)), 0, 1), in place
  cumsum_kernel<<<(B_ * NUMC_ + 255) / 256, 256, 0, stream>>>(o_kc, P(89));
  // m_in = cat[h1, v1, ssk] padded 513 -> 544
  cat_m_kernel<<<4096, 256, 0, stream>>>(ctx, val, o_kc, qry, xbf);
  convW(P(90), 513, 512, 544);
  gemm(xbf, 544, P(91), nullptr, ybf, 512, 544, 1);
  convW(P(92), 512, 256, 512);
  gemm(ybf, 512, P(93), nullptr, qbf, 256, 512, 1);
  dot_head_kernel<<<M_ / 8, 256, 0, stream>>>(qbf, P(94), P(95), o_mast, o_mlogit);
}
